// multi_head_attention_61143154426271
// MI455X (gfx1250) — matrix operand order repaired and checked by reading, not compiled, not run on hardware
//
#include <hip/hip_runtime.h>
#include <hip/hip_bf16.h>

typedef __attribute__((ext_vector_type(16))) _Float16 v16h;
typedef __attribute__((ext_vector_type(8)))  float    v8f;
typedef __attribute__((ext_vector_type(4)))  unsigned u32x4;
typedef __attribute__((ext_vector_type(4)))  float    f32x4;

#define DMODEL 1024
#define NHEAD  16
#define HDIM   64
#define BATCH  2
#define SEQ    2048
#define ROWS   (BATCH * SEQ)   // 4096

union Op {
    v16h  v;
    unsigned u[8];
    u32x4 q[2];
};

__device__ __forceinline__ unsigned pk2(float a, float b) {
    union { _Float16 h[2]; unsigned u; } x;
    x.h[0] = (_Float16)a; x.h[1] = (_Float16)b;
    return x.u;
}
__device__ __forceinline__ unsigned short h1(float a) {
    union { _Float16 h; unsigned short u; } x;
    x.h = (_Float16)a;
    return x.u;
}

// Cross-half-wave (lane ^ 16) exchange via v_permlanex16_b32 (pure VALU,
// avoids ds_bpermute + s_wait_dscnt round trips).
__device__ __forceinline__ unsigned x16u(unsigned x) {
    return (unsigned)__builtin_amdgcn_permlanex16(
        (int)x, (int)x, 0x76543210, 0xfedcba98, false, false);
}
__device__ __forceinline__ float x16f(float x) {
    int i = __builtin_bit_cast(int, x);
    int r = __builtin_amdgcn_permlanex16(i, i, 0x76543210, 0xfedcba98, false, false);
    return __builtin_bit_cast(float, r);
}

// ---------------------------------------------------------------------------
// Convert fp32 weight W[k][n] -> f16 transposed Wt[n][k] (B-operand friendly)
// ---------------------------------------------------------------------------
__global__ __launch_bounds__(256) void wt_prep(const float* __restrict__ W,
                                               unsigned short* __restrict__ Wt) {
    int idx = blockIdx.x * 256 + threadIdx.x;      // 0 .. 1024*1024-1
    int n = idx >> 10, k = idx & 1023;
    Wt[n * DMODEL + k] = h1(W[k * DMODEL + n]);
}

// ---------------------------------------------------------------------------
// GEMM: C[4096x1024] = A[4096x1024] * W[1024x1024] (+bias)(+residual)
//   MODE 0: A fp32, out f16 row-major                (Q,K projections)
//   MODE 1: A fp32, out f16 transposed-per-head      (V projection -> Vt)
//   MODE 2: A f16,  out fp32 row-major + residual    (output projection)
// Each wave computes a 32x64 tile (2x4 WMMA D-tiles), K-step = 32.
// ---------------------------------------------------------------------------
template <int MODE>
__global__ __launch_bounds__(256) void gemm_proj(
    const float* __restrict__ Af32, const unsigned short* __restrict__ Ah,
    const unsigned short* __restrict__ Wt, const float* __restrict__ bias,
    const float* __restrict__ resid, unsigned short* __restrict__ outh,
    float* __restrict__ outf)
{
    const int lane = threadIdx.x & 31;
    const int wave = threadIdx.x >> 5;
    const int w    = blockIdx.x * 8 + wave;        // 2048 waves total
    const int wm   = w & 127;                      // 128 M-tiles of 32
    const int wn   = w >> 7;                       // 16  N-tiles of 64
    const int m0   = wm * 32;
    const int n0   = wn * 64;
    const int lm   = lane & 15;
    const int hh   = lane >> 4;

    v8f acc[2][4];
    const v8f vzero = {0.f,0.f,0.f,0.f,0.f,0.f,0.f,0.f};
#pragma unroll
    for (int t = 0; t < 2; ++t)
#pragma unroll
        for (int u = 0; u < 4; ++u) acc[t][u] = vzero;

    for (int k0 = 0; k0 < DMODEL; k0 += 32) {
        Op a[2];
#pragma unroll
        for (int t = 0; t < 2; ++t) {
            const int row = m0 + 16 * t + lm;
            if (MODE == 2) {
                const unsigned short* p = Ah + row * DMODEL + k0 + 8 * hh;
                a[t].q[0] = *(const u32x4*)p;
                a[t].q[1] = *(const u32x4*)(p + 16);
            } else {
                const float* p = Af32 + row * DMODEL + k0 + 8 * hh;
                f32x4 x0 = *(const f32x4*)p;
                f32x4 x1 = *(const f32x4*)(p + 4);
                f32x4 y0 = *(const f32x4*)(p + 16);
                f32x4 y1 = *(const f32x4*)(p + 20);
                a[t].u[0] = pk2(x0.x, x0.y); a[t].u[1] = pk2(x0.z, x0.w);
                a[t].u[2] = pk2(x1.x, x1.y); a[t].u[3] = pk2(x1.z, x1.w);
                a[t].u[4] = pk2(y0.x, y0.y); a[t].u[5] = pk2(y0.z, y0.w);
                a[t].u[6] = pk2(y1.x, y1.y); a[t].u[7] = pk2(y1.z, y1.w);
            }
        }
        Op bb[4];
#pragma unroll
        for (int u = 0; u < 4; ++u) {
            const unsigned short* p = Wt + (n0 + 16 * u + lm) * DMODEL + k0 + 8 * hh;
            bb[u].q[0] = *(const u32x4*)p;
            bb[u].q[1] = *(const u32x4*)(p + 16);
        }
#pragma unroll
        for (int t = 0; t < 2; ++t)
#pragma unroll
            for (int u = 0; u < 4; ++u)
                acc[t][u] = __builtin_amdgcn_wmma_f32_16x16x32_f16(
                    false, a[t].v, false, bb[u].v, (short)0, acc[t][u], false, false);
    }

#pragma unroll
    for (int t = 0; t < 2; ++t)
#pragma unroll
        for (int u = 0; u < 4; ++u) {
            const int col  = n0 + 16 * u + lm;
            const float bc = bias[col];
#pragma unroll
            for (int v = 0; v < 8; ++v) {
                const int   row = m0 + 16 * t + v + 8 * hh;
                const float val = acc[t][u][v] + bc;
                if (MODE == 0) {
                    outh[row * DMODEL + col] = h1(val);
                } else if (MODE == 1) {
                    const int b = row >> 11, s = row & 2047;
                    const int hd = col >> 6, dd = col & 63;
                    outh[((b * NHEAD + hd) * HDIM + dd) * SEQ + s] = h1(val);
                } else {
                    outf[row * DMODEL + col] = val + resid[row * DMODEL + col];
                }
            }
        }
}

// ---------------------------------------------------------------------------
// Flash attention: one wave per 16-query tile of one (b,h).
// S^T = K*Q^T, online softmax over key rows, ctx^T = V^T * P^T.
// ---------------------------------------------------------------------------
__global__ __launch_bounds__(256) void attn_kernel(
    const unsigned short* __restrict__ Qp, const unsigned short* __restrict__ Kp,
    const unsigned short* __restrict__ Vt, unsigned short* __restrict__ ctx)
{
    const int lane = threadIdx.x & 31;
    const int wave = threadIdx.x >> 5;
    const int T    = blockIdx.x * 8 + wave;        // 4096 q-tiles
    const int qt   = T & 127;
    const int hd   = (T >> 7) & 15;
    const int b    = T >> 11;
    const int lm   = lane & 15;
    const int hh   = lane >> 4;

    const int qrow = b * SEQ + qt * 16 + lm;       // this lane's query row
    const unsigned short* qb = Qp + qrow * DMODEL + hd * HDIM;

    Op qop[2];
#pragma unroll
    for (int s = 0; s < 2; ++s) {
        const unsigned short* p = qb + 32 * s + 8 * hh;
        qop[s].q[0] = *(const u32x4*)p;
        qop[s].q[1] = *(const u32x4*)(p + 16);
    }

    const v8f vzero = {0.f,0.f,0.f,0.f,0.f,0.f,0.f,0.f};
    v8f acc[4];
#pragma unroll
    for (int c = 0; c < 4; ++c) acc[c] = vzero;

    float m_run = -1e30f, l_run = 0.f;
    const float scale = 0.125f;                    // 1/sqrt(64)

    for (int kb = 0; kb < SEQ; kb += 32) {
        v8f st[2];
#pragma unroll
        for (int t = 0; t < 2; ++t) {
            st[t] = vzero;
            const unsigned short* kr =
                Kp + (b * SEQ + kb + 16 * t + lm) * DMODEL + hd * HDIM;
#pragma unroll
            for (int s = 0; s < 2; ++s) {
                Op ak;
                const unsigned short* p = kr + 32 * s + 8 * hh;
                ak.q[0] = *(const u32x4*)p;
                ak.q[1] = *(const u32x4*)(p + 16);
                st[t] = __builtin_amdgcn_wmma_f32_16x16x32_f16(
                    false, ak.v, false, qop[s].v, (short)0, st[t], false, false);
            }
        }
        // --- online softmax over key rows of this 32-key block ---
        float tmax = -1e30f;
#pragma unroll
        for (int v = 0; v < 8; ++v)
            tmax = fmaxf(tmax, fmaxf(st[0][v], st[1][v]));
        tmax *= scale;
        tmax = fmaxf(tmax, x16f(tmax));
        const float m_new = fmaxf(m_run, tmax);
        const float corr  = __expf(m_run - m_new);

        float e0[8], e1[8];
        float ssum = 0.f;
#pragma unroll
        for (int v = 0; v < 8; ++v) {
            e0[v] = __expf(st[0][v] * scale - m_new);
            e1[v] = __expf(st[1][v] * scale - m_new);
            ssum += e0[v] + e1[v];
        }
        ssum += x16f(ssum);
        l_run = l_run * corr + ssum;
        m_run = m_new;

        // Build P^T B-operand in registers: pack exp pairs to f16x2 first,
        // then a single cross-half permlanex16 per packed dword (8 total).
        Op pop;
#pragma unroll
        for (int j = 0; j < 4; ++j) {
            const unsigned pe0 = pk2(e0[2*j], e0[2*j+1]);   // own keys (tile 0)
            const unsigned pe1 = pk2(e1[2*j], e1[2*j+1]);   // own keys (tile 1)
            const unsigned ox0 = x16u(pe0);                  // partner tile-0 pairs
            const unsigned ox1 = x16u(pe1);                  // partner tile-1 pairs
            pop.u[j]     = (hh == 0) ? pe0 : ox1;
            pop.u[4 + j] = (hh == 0) ? ox0 : pe1;
        }

#pragma unroll
        for (int c = 0; c < 4; ++c)
#pragma unroll
            for (int v = 0; v < 8; ++v) acc[c][v] *= corr;

        // ctx^T += V^T * P^T   (A = Vt rows contiguous over keys)
#pragma unroll
        for (int c = 0; c < 4; ++c) {
            Op av;
            const unsigned short* vp =
                Vt + ((b * NHEAD + hd) * HDIM + 16 * c + lm) * SEQ + kb + 16 * hh;
            av.q[0] = *(const u32x4*)vp;
            av.q[1] = *(const u32x4*)(vp + 8);
            acc[c] = __builtin_amdgcn_wmma_f32_16x16x32_f16(
                false, av.v, false, pop.v, (short)0, acc[c], false, false);
        }
    }

    const float inv = 1.f / l_run;
#pragma unroll
    for (int c = 0; c < 4; ++c)
#pragma unroll
        for (int v = 0; v < 8; ++v) {
            const int n = 16 * c + v + 8 * hh;     // head depth index
            ctx[qrow * DMODEL + hd * HDIM + n] = h1(acc[c][v] * inv);
        }
}

// ---------------------------------------------------------------------------
// LayerNorm over last axis (1024), one block (8 waves) per row.
// ---------------------------------------------------------------------------
__global__ __launch_bounds__(256) void layernorm_kernel(
    const float* __restrict__ X, const float* __restrict__ gamma,
    const float* __restrict__ beta, float* __restrict__ out)
{
    __shared__ float s1[8], s2[8];
    const int r = blockIdx.x;
    const float* x = X + r * DMODEL;

    float vals[4];
    float sum = 0.f, sq = 0.f;
#pragma unroll
    for (int j = 0; j < 4; ++j) {
        const float v = x[threadIdx.x + 256 * j];
        vals[j] = v; sum += v; sq += v * v;
    }
#pragma unroll
    for (int o = 16; o >= 1; o >>= 1) {
        sum += __shfl_xor(sum, o, 32);
        sq  += __shfl_xor(sq,  o, 32);
    }
    const int lane = threadIdx.x & 31, wv = threadIdx.x >> 5;
    if (lane == 0) { s1[wv] = sum; s2[wv] = sq; }
    __syncthreads();
    if (wv == 0) {
        float a = (lane < 8) ? s1[lane] : 0.f;
        float c = (lane < 8) ? s2[lane] : 0.f;
#pragma unroll
        for (int o = 4; o >= 1; o >>= 1) {
            a += __shfl_xor(a, o, 32);
            c += __shfl_xor(c, o, 32);
        }
        if (lane == 0) { s1[0] = a; s2[0] = c; }
    }
    __syncthreads();
    const float mu   = s1[0] * (1.f / DMODEL);
    const float var  = s2[0] * (1.f / DMODEL) - mu * mu;
    const float rstd = rsqrtf(var + 1e-6f);
#pragma unroll
    for (int j = 0; j < 4; ++j) {
        const int i = threadIdx.x + 256 * j;
        out[r * DMODEL + i] = gamma[i] * (vals[j] - mu) * rstd + beta[i];
    }
}

// ---------------------------------------------------------------------------
extern "C" void kernel_launch(void* const* d_in, const int* in_sizes, int n_in,
                              void* d_out, int out_size, void* d_ws, size_t ws_size,
                              hipStream_t stream) {
    (void)in_sizes; (void)n_in; (void)out_size; (void)ws_size;

    const float* v_in  = (const float*)d_in[0];
    const float* k_in  = (const float*)d_in[1];
    const float* q_in  = (const float*)d_in[2];
    const float* Wv    = (const float*)d_in[3];
    const float* bv    = (const float*)d_in[4];
    const float* Wk    = (const float*)d_in[5];
    const float* bk    = (const float*)d_in[6];
    const float* Wq    = (const float*)d_in[7];
    const float* bq    = (const float*)d_in[8];
    const float* Wo    = (const float*)d_in[9];
    const float* bo    = (const float*)d_in[10];
    const float* gamma = (const float*)d_in[11];
    const float* beta  = (const float*)d_in[12];

    char* ws = (char*)d_ws;
    const size_t WSZ = (size_t)DMODEL * DMODEL * 2;       // 2 MB per f16 weight
    const size_t MSZ = (size_t)ROWS * DMODEL * 2;         // 8 MB per f16 activation
    unsigned short* WtQ = (unsigned short*)(ws + 0 * WSZ);
    unsigned short* WtK = (unsigned short*)(ws + 1 * WSZ);
    unsigned short* WtV = (unsigned short*)(ws + 2 * WSZ);
    unsigned short* WtO = (unsigned short*)(ws + 3 * WSZ);
    unsigned short* Qp  = (unsigned short*)(ws + 4 * WSZ);
    unsigned short* Kp  = (unsigned short*)(ws + 4 * WSZ + 1 * MSZ);
    unsigned short* Vt  = (unsigned short*)(ws + 4 * WSZ + 2 * MSZ);
    unsigned short* Ctx = (unsigned short*)(ws + 4 * WSZ + 3 * MSZ);
    float*          Of  = (float*)         (ws + 4 * WSZ + 4 * MSZ);   // 16 MB fp32

    // 1) weights -> transposed f16
    wt_prep<<<4096, 256, 0, stream>>>(Wq, WtQ);
    wt_prep<<<4096, 256, 0, stream>>>(Wk, WtK);
    wt_prep<<<4096, 256, 0, stream>>>(Wv, WtV);
    wt_prep<<<4096, 256, 0, stream>>>(Wo, WtO);

    // 2) projections (2048 waves -> 256 blocks of 8 waves)
    gemm_proj<0><<<256, 256, 0, stream>>>(q_in, nullptr, WtQ, bq, nullptr, Qp, nullptr);
    gemm_proj<0><<<256, 256, 0, stream>>>(k_in, nullptr, WtK, bk, nullptr, Kp, nullptr);
    gemm_proj<1><<<256, 256, 0, stream>>>(v_in, nullptr, WtV, bv, nullptr, Vt, nullptr);

    // 3) attention: 4096 q-tiles -> 512 blocks of 8 waves
    attn_kernel<<<512, 256, 0, stream>>>(Qp, Kp, Vt, Ctx);

    // 4) output projection + bias + residual (fp32)
    gemm_proj<2><<<256, 256, 0, stream>>>(nullptr, Ctx, WtO, bo, q_in, nullptr, Of);

    // 5) LayerNorm -> d_out
    layernorm_kernel<<<4096, 256, 0, stream>>>(Of, gamma, beta, (float*)d_out);
}
